// Net4_4217657885329
// MI455X (gfx1250) — 8-bit matrix operand order repaired and checked by reading, compiled, not run on hardware
//
#include <hip/hip_runtime.h>
#include <hip/hip_bf16.h>

#define CIN   256
#define E_    16
#define L_    64
#define M_    64
#define H_    64
#define COUT  256
#define B_    16
#define N_    2048
#define NP    (N_ + L_)
#define D2    1024

typedef _Float16 v16h __attribute__((ext_vector_type(16)));
typedef _Float16 v8h  __attribute__((ext_vector_type(8)));
typedef float    v8f  __attribute__((ext_vector_type(8)));
typedef int      v8i  __attribute__((ext_vector_type(8)));

#define OFF_ESEQ   0u
#define OFF_W2TW   1081344u
#define OFF_W2TB   1212416u
#define OFF_W1TW   1343488u
#define OFF_W1TB   1351680u
#define OFF_BDT    1359872u
#define OFF_DT     1368064u
#define OFF_WDBT   1400832u
#define OFF_WDEC8  1409024u
#define OFF_PREW   1671168u
#define OFF_PREB   10059776u
#define OFF_MHIST  18448384u

__device__ inline unsigned char f32_to_e4m3(float f) {
  unsigned int u = __float_as_uint(f);
  unsigned int s = (u >> 24) & 0x80u;
  float a = fabsf(f);
  if (!(a < 448.0f)) return (unsigned char)(s | 0x7Eu);
  if (a < 0.0009765625f) return (unsigned char)s;
  int e; (void)frexpf(a, &e); int E = e - 1;
  if (E < -6) E = -6;
  int qi = (int)rintf(a * exp2f((float)(3 - E)));
  if (qi >= 16) { E += 1; qi = 8; }
  unsigned int bits = (qi < 8) ? (unsigned)qi : (unsigned)(((E + 7) << 3) | (qi - 8));
  return (unsigned char)(s | bits);
}

__device__ inline v16h pack16(v8h lo, v8h hi) {
  v16h r;
  for (int i = 0; i < 8; ++i) { r[i] = lo[i]; r[i + 8] = hi[i]; }
  return r;
}

__global__ void prep_kernel(const int* __restrict__ x0, const float* __restrict__ emb,
                            const float* __restrict__ Wencw, const float* __restrict__ bencw,
                            const float* __restrict__ Wdecw, const float* __restrict__ Wdecb,
                            const float* __restrict__ bdecw, const float* __restrict__ decw,
                            _Float16* eseq, _Float16* w2TW, _Float16* w2TB,
                            _Float16* w1TW, _Float16* w1TB, _Float16* bdT,
                            _Float16* dT, _Float16* wdbT, unsigned char* wdec8)
{
  const int S0 = B_ * NP * E_;
  const int S1 = 65536, S3 = 4096, S6 = 16384, S7 = 4096 * 64;
  const int total = S0 + 2 * S1 + 3 * S3 + S6 + S3 + S7;
  for (int id = blockIdx.x * blockDim.x + threadIdx.x; id < total;
       id += gridDim.x * blockDim.x) {
    int t = id;
    if (t < S0) {
      int b = t / (NP * E_); int r = t - b * NP * E_;
      int p = r >> 4, e = r & 15;
      int tok = (p < L_) ? 0 : x0[b * N_ + (p - L_)];
      eseq[t] = (_Float16)emb[tok * E_ + e];
      continue;
    }
    t -= S0;
    if (t < S1) { int h = t >> 10, d = t & 1023; w2TW[t] = (_Float16)Wencw[(M_ + d) * H_ + h]; continue; }
    t -= S1;
    if (t < S1) { int h = t >> 10, d = t & 1023; w2TB[t] = (_Float16)bencw[(M_ + d) * H_ + h]; continue; }
    t -= S1;
    if (t < S3) { int h = t >> 6, j = t & 63; w1TW[t] = (_Float16)Wencw[j * H_ + h]; continue; }
    t -= S3;
    if (t < S3) { int h = t >> 6, j = t & 63; w1TB[t] = (_Float16)bencw[j * H_ + h]; continue; }
    t -= S3;
    if (t < S3) { int i = t >> 6, k = t & 63; bdT[t] = (_Float16)bdecw[k * M_ + i]; continue; }
    t -= S3;
    if (t < S6) { int c = t >> 6, k = t & 63; dT[t] = (_Float16)decw[k * COUT + c]; continue; }
    t -= S6;
    if (t < S3) { wdbT[t] = (_Float16)Wdecb[t]; continue; }
    t -= S3;
    wdec8[t] = f32_to_e4m3(Wdecw[t]);
  }
}

__global__ __launch_bounds__(256) void
pre_gemm_kernel(const _Float16* __restrict__ eseq,
                const _Float16* __restrict__ w2TW, const _Float16* __restrict__ w2TB,
                const float* __restrict__ Wencb, const float* __restrict__ bencb,
                float* __restrict__ preW, float* __restrict__ preB)
{
  int gw = blockIdx.x * (blockDim.x >> 5) + (threadIdx.x >> 5);
  int lane = threadIdx.x & 31, lr = lane & 15, sel = lane >> 4;
  int ttile = gw & 127; int rest = gw >> 7;
  int htile = rest & 3; rest >>= 2;
  int b = rest & 15; int enc = rest >> 4;
  const _Float16* Bmat = enc ? w2TB : w2TW;
  const float* bias = enc ? bencb : Wencb;
  float* out = enc ? preB : preW;
  int t0 = ttile * 16, h0 = htile * 16;
  const _Float16* arow = eseq + (size_t)b * (NP * E_) + (size_t)(t0 + lr) * E_;
  const _Float16* bcol = Bmat + (size_t)(h0 + lr) * D2;
  v8f c = {};
  for (int ks = 0; ks < 32; ++ks) {
    int kk = ks * 32;
    v8h alo = *(const v8h*)(arow + kk + sel * 8);
    v8h ahi = *(const v8h*)(arow + kk + 16 + sel * 8);
    v8h blo = *(const v8h*)(bcol + kk + sel * 8);
    v8h bhi = *(const v8h*)(bcol + kk + 16 + sel * 8);
    c = __builtin_amdgcn_wmma_f32_16x16x32_f16(false, pack16(alo, ahi),
                                               false, pack16(blo, bhi),
                                               (short)0, c, false, false);
  }
  float bb = bias[h0 + lr];
  for (int v = 0; v < 8; ++v) {
    int tt = t0 + v + 8 * sel;
    out[((size_t)tt * B_ + b) * H_ + (h0 + lr)] = c[v] + bb;
  }
}

#define SCAN_LDS 317184

__global__ __launch_bounds__(512, 1) void
scan_kernel(const unsigned char* __restrict__ wdec8_g,
            const _Float16* __restrict__ w1TW_g, const _Float16* __restrict__ w1TB_g,
            const _Float16* __restrict__ bdT_g, const _Float16* __restrict__ wdbT_g,
            const float* __restrict__ Wencb_g, const float* __restrict__ bencb_g,
            const float* __restrict__ bdecb_g,
            const float* __restrict__ preW, const float* __restrict__ preB,
            float* __restrict__ mhist)
{
  extern __shared__ char smem[];
  unsigned int* wd8  = (unsigned int*)(smem);
  _Float16* w1T      = (_Float16*)(smem + 262144);
  _Float16* bdT      = (_Float16*)(smem + 278528);
  _Float16* wdbT     = (_Float16*)(smem + 286720);
  float* m_lds       = (float*)(smem + 294912);
  unsigned int* mfp8 = (unsigned int*)(smem + 299008);
  float* hW          = (float*)(smem + 300032);
  float* hB          = (float*)(smem + 304128);
  float* z           = (float*)(smem + 308224);
  float* bmv         = (float*)(smem + 312320);
  float* encb        = (float*)(smem + 316416);
  float* bdecb       = (float*)(smem + 316928);

  int tid = threadIdx.x, w = tid >> 5, lane = tid & 31, lr = lane & 15, sel = lane >> 4;

  const unsigned int* wd8g = (const unsigned int*)wdec8_g;
  for (int i = tid; i < 65536; i += 512) wd8[i] = wd8g[i];
  for (int i = tid; i < 4096; i += 512) {
    w1T[i] = w1TW_g[i]; w1T[4096 + i] = w1TB_g[i];
    bdT[i] = bdT_g[i];  wdbT[i] = wdbT_g[i];
  }
  for (int i = tid; i < 1024; i += 512) { m_lds[i] = 0.f; z[i] = 0.f; }
  for (int i = tid; i < 256; i += 512) mfp8[i] = 0u;
  if (tid < 64) { encb[tid] = Wencb_g[tid]; encb[64 + tid] = bencb_g[tid]; bdecb[tid] = bdecb_g[tid]; }
  __syncthreads();

  for (int t = 0; t < N_; ++t) {
    if (w < 8) {
      int enc = w >> 2, nt = w & 3, h0 = nt * 16;
      const _Float16* BT = w1T + enc * 4096;
      const float* preX = (enc ? preB : preW) + (size_t)t * (B_ * H_);
      v8f c = {};
      for (int ks = 0; ks < 2; ++ks) {
        int kk = ks * 32;
        v16h a;
        for (int i = 0; i < 8; ++i) {
          a[i]     = (_Float16)m_lds[lr * 64 + kk + sel * 8 + i];
          a[i + 8] = (_Float16)m_lds[lr * 64 + kk + 16 + sel * 8 + i];
        }
        const _Float16* bc = BT + (h0 + lr) * 64;
        v8h blo = *(const v8h*)(bc + kk + sel * 8);
        v8h bhi = *(const v8h*)(bc + kk + 16 + sel * 8);
        c = __builtin_amdgcn_wmma_f32_16x16x32_f16(false, a, false, pack16(blo, bhi),
                                                   (short)0, c, false, false);
      }
      float* hout = enc ? hB : hW;
      float bb = encb[enc * 64 + h0 + lr];
      for (int v = 0; v < 8; ++v) {
        int b = v + 8 * sel;
        float x = c[v] + preX[b * H_ + (h0 + lr)] + bb;
        hout[b * 64 + h0 + lr] = 1.0f / (1.0f + __expf(-x));
      }
    }
    if (t + 1 < N_) {
      __builtin_prefetch(preW + (size_t)(t + 1) * (B_ * H_) + tid * 2, 0, 1);
      __builtin_prefetch(preB + (size_t)(t + 1) * (B_ * H_) + tid * 2, 0, 1);
    }
    __syncthreads();

    if (w < 4) {
      int i0 = w * 16;
      v8f c = {};
      for (int ks = 0; ks < 2; ++ks) {
        int kk = ks * 32;
        v16h a;
        for (int i = 0; i < 8; ++i) {
          a[i]     = (_Float16)hB[lr * 64 + kk + sel * 8 + i];
          a[i + 8] = (_Float16)hB[lr * 64 + kk + 16 + sel * 8 + i];
        }
        const _Float16* bc = bdT + (i0 + lr) * 64;
        v8h blo = *(const v8h*)(bc + kk + sel * 8);
        v8h bhi = *(const v8h*)(bc + kk + 16 + sel * 8);
        c = __builtin_amdgcn_wmma_f32_16x16x32_f16(false, a, false, pack16(blo, bhi),
                                                   (short)0, c, false, false);
      }
      for (int v = 0; v < 8; ++v)
        bmv[(v + 8 * sel) * 64 + i0 + lr] = c[v] + bdecb[i0 + lr];
    } else if (w < 8) {
      int i0 = (w - 4) * 16;
      v8f c = {};
      for (int ks = 0; ks < 2; ++ks) {
        int kk = ks * 32;
        v16h a;
        for (int i = 0; i < 8; ++i) {
          a[i]     = (_Float16)m_lds[lr * 64 + kk + sel * 8 + i];
          a[i + 8] = (_Float16)m_lds[lr * 64 + kk + 16 + sel * 8 + i];
        }
        const _Float16* bc = wdbT + (i0 + lr) * 64;
        v8h blo = *(const v8h*)(bc + kk + sel * 8);
        v8h bhi = *(const v8h*)(bc + kk + 16 + sel * 8);
        c = __builtin_amdgcn_wmma_f32_16x16x32_f16(false, a, false, pack16(blo, bhi),
                                                   (short)0, c, false, false);
      }
      for (int v = 0; v < 8; ++v)
        atomicAdd(&z[(v + 8 * sel) * 64 + i0 + lr], c[v]);
    }

    {
      int it = w & 3, i0 = it * 16, kb = (w >> 2) * 16;
      v8i bfrag;
      for (int v = 0; v < 8; ++v)
        bfrag[v] = (int)mfp8[lr * 16 + sel * 2 + (v >> 1) * 4 + (v & 1)];
      v8f facc = {};
      for (int k = kb; k < kb + 16; ++k) {
        int r0 = k * 64 + i0;
        v8i afrag;
        for (int v = 0; v < 8; ++v)
          afrag[v] = (int)wd8[(r0 + lr) * 16 + sel * 2 + (v >> 1) * 4 + (v & 1)];
        v8f zero = {};
        v8f d = __builtin_amdgcn_wmma_f32_16x16x64_fp8_fp8(afrag, bfrag, (short)0,
                                                           zero, false, false);
        float hv = hW[lr * 64 + k];
        for (int v = 0; v < 8; ++v) facc[v] += hv * d[v];
      }
      for (int v = 0; v < 8; ++v)
        atomicAdd(&z[lr * 64 + i0 + v + 8 * sel], facc[v]);
    }
    __syncthreads();

    float* mh = mhist + (size_t)t * (B_ * M_);
    for (int idx = tid; idx < 1024; idx += 512) {
      float val = 1.0f / (1.0f + __expf(-(z[idx] + bmv[idx])));
      m_lds[idx] = val;
      mh[idx] = val;
      z[idx] = 0.f;
    }
    __syncthreads();
    if (tid < 256) {
      int base = tid * 4;
      unsigned int p = (unsigned)f32_to_e4m3(m_lds[base])
                     | ((unsigned)f32_to_e4m3(m_lds[base + 1]) << 8)
                     | ((unsigned)f32_to_e4m3(m_lds[base + 2]) << 16)
                     | ((unsigned)f32_to_e4m3(m_lds[base + 3]) << 24);
      mfp8[tid] = p;
    }
    __syncthreads();
  }
}

__global__ __launch_bounds__(128) void
loss_kernel(const float* __restrict__ mhist, const _Float16* __restrict__ dT,
            const float* __restrict__ decb, const int* __restrict__ x0,
            float* __restrict__ out)
{
  __shared__ float m_s[B_ * M_];
  __shared__ float lg[B_ * COUT];
  int t = blockIdx.x, tid = threadIdx.x;
  int w = tid >> 5, lane = tid & 31, lr = lane & 15, sel = lane >> 4;
  for (int i = tid; i < B_ * M_; i += 128) m_s[i] = mhist[(size_t)t * (B_ * M_) + i];
  __syncthreads();
  for (int q = 0; q < 4; ++q) {
    int c0 = (w * 4 + q) * 16;
    v8f c = {};
    for (int ks = 0; ks < 2; ++ks) {
      int kk = ks * 32;
      v16h a;
      for (int i = 0; i < 8; ++i) {
        a[i]     = (_Float16)m_s[lr * 64 + kk + sel * 8 + i];
        a[i + 8] = (_Float16)m_s[lr * 64 + kk + 16 + sel * 8 + i];
      }
      const _Float16* bc = dT + (size_t)(c0 + lr) * 64;
      v8h blo = *(const v8h*)(bc + kk + sel * 8);
      v8h bhi = *(const v8h*)(bc + kk + 16 + sel * 8);
      c = __builtin_amdgcn_wmma_f32_16x16x32_f16(false, a, false, pack16(blo, bhi),
                                                 (short)0, c, false, false);
    }
    float bb = decb[c0 + lr];
    for (int v = 0; v < 8; ++v)
      lg[(v + 8 * sel) * COUT + c0 + lr] = c[v] + bb;
  }
  __syncthreads();
  if (tid < 16) {
    int b = tid;
    const float* row = lg + b * COUT;
    float mx = row[0];
    for (int i = 1; i < COUT; ++i) mx = fmaxf(mx, row[i]);
    float s = 0.f;
    for (int i = 0; i < COUT; ++i) s += __expf(row[i] - mx);
    float lse = mx + __logf(s);
    int y = x0[b * N_ + t];
    out[t * B_ + b] = (lse - row[y]) * 1.44269504088896340736f;
  }
}

extern "C" void kernel_launch(void* const* d_in, const int* in_sizes, int n_in,
                              void* d_out, int out_size, void* d_ws, size_t ws_size,
                              hipStream_t stream) {
  const int*   x0    = (const int*)  d_in[0];
  const float* emb   = (const float*)d_in[1];
  const float* Wencw = (const float*)d_in[2];
  const float* Wencb = (const float*)d_in[3];
  const float* Wdecw = (const float*)d_in[4];
  const float* Wdecb = (const float*)d_in[5];
  const float* bencw = (const float*)d_in[6];
  const float* bencb = (const float*)d_in[7];
  const float* bdecw = (const float*)d_in[8];
  const float* bdecb = (const float*)d_in[9];
  const float* decw  = (const float*)d_in[10];
  const float* decb  = (const float*)d_in[11];
  float* out = (float*)d_out;
  char* ws = (char*)d_ws;

  _Float16* eseq  = (_Float16*)(ws + OFF_ESEQ);
  _Float16* w2TW  = (_Float16*)(ws + OFF_W2TW);
  _Float16* w2TB  = (_Float16*)(ws + OFF_W2TB);
  _Float16* w1TW  = (_Float16*)(ws + OFF_W1TW);
  _Float16* w1TB  = (_Float16*)(ws + OFF_W1TB);
  _Float16* bdT   = (_Float16*)(ws + OFF_BDT);
  _Float16* dT    = (_Float16*)(ws + OFF_DT);
  _Float16* wdbT  = (_Float16*)(ws + OFF_WDBT);
  unsigned char* wdec8 = (unsigned char*)(ws + OFF_WDEC8);
  float* preW  = (float*)(ws + OFF_PREW);
  float* preB  = (float*)(ws + OFF_PREB);
  float* mhist = (float*)(ws + OFF_MHIST);

  (void)hipFuncSetAttribute((const void*)scan_kernel,
                            hipFuncAttributeMaxDynamicSharedMemorySize, SCAN_LDS);

  prep_kernel<<<1024, 256, 0, stream>>>(x0, emb, Wencw, bencw, Wdecw, Wdecb,
                                        bdecw, decw, eseq, w2TW, w2TB,
                                        w1TW, w1TB, bdT, dT, wdbT, wdec8);

  pre_gemm_kernel<<<2048, 256, 0, stream>>>(eseq, w2TW, w2TB, Wencb, bencb,
                                            preW, preB);

  scan_kernel<<<1, 512, SCAN_LDS, stream>>>(wdec8, w1TW, w1TB, bdT, wdbT,
                                            Wencb, bencb, bdecb,
                                            preW, preB, mhist);

  loss_kernel<<<2048, 128, 0, stream>>>(mhist, dT, decb, x0, out);
}
